// S5Layer_26645977105183
// MI455X (gfx1250) — matrix operand order repaired and checked by reading, not compiled, not run on hardware
//
#include <hip/hip_runtime.h>
#include <math.h>
#include <stdint.h>

typedef __attribute__((ext_vector_type(16))) __bf16 v16bf;
typedef __attribute__((ext_vector_type(8)))  float  v8f;

constexpr int Bsz = 8, Lsz = 2048, Dsz = 1024, Psz = 256;
constexpr int Msz = Bsz * Lsz;            // 16384 rows
constexpr int NC  = 16, CH = Lsz / NC;    // scan chunks: 16 x 128

// ---------- bf16 helpers (bit-exact, no reliance on native conversions) ----
__device__ inline __bf16 f2bf(float f) {
  union { float f; unsigned u; } in; in.f = f;
  unsigned u = in.u;
  u += 0x7fffu + ((u >> 16) & 1u);              // round-to-nearest-even
  union { unsigned short s; __bf16 b; } out;
  out.s = (unsigned short)(u >> 16);
  return out.b;
}
__device__ inline float bf2f(__bf16 b) {
  union { __bf16 b; unsigned short s; } in; in.b = b;
  union { unsigned u; float f; } out;
  out.u = ((unsigned)in.s) << 16;
  return out.f;
}
__device__ inline float2 cmul(float2 a, float2 b) {
  return make_float2(a.x*b.x - a.y*b.y, a.x*b.y + a.y*b.x);
}

// ---------- small dense matmul: out[8,N] = t[8,K] @ W[K,N] + bias ----------
__global__ __launch_bounds__(256) void small_mm_kernel(
    const float* __restrict__ t, const float* __restrict__ W,
    const float* __restrict__ bias, float* __restrict__ out, int N, int K) {
  int idx = blockIdx.x * 256 + threadIdx.x;
  int row = idx / N, col = idx % N;
  float acc = bias[col];
  const float* tr = t + (size_t)row * K;
  for (int k = 0; k < K; k++) acc += tr[k] * W[(size_t)k * N + col];
  out[idx] = acc;
}

// ---------- transpose f32 (K,N) -> bf16 (N,K) ------------------------------
__global__ __launch_bounds__(256) void transpose_bf16_kernel(
    const float* __restrict__ W, __bf16* __restrict__ WT, int K, int N) {
  __shared__ float tile[32][33];
  int n0 = blockIdx.x * 32, k0 = blockIdx.y * 32;
  int tx = threadIdx.x, ty = threadIdx.y;            // (32,8)
  #pragma unroll
  for (int j = 0; j < 32; j += 8)
    tile[ty + j][tx] = W[(size_t)(k0 + ty + j) * N + n0 + tx];
  __syncthreads();
  #pragma unroll
  for (int j = 0; j < 32; j += 8)
    WT[(size_t)(n0 + ty + j) * K + k0 + tx] = f2bf(tile[tx][ty + j]);
}

// ---------- ZOH discretization: B_bar (re/im interleaved rows, bf16) -------
__global__ __launch_bounds__(256) void prep_bmat_kernel(
    const float* __restrict__ Lre, const float* __restrict__ Lim,
    const float* __restrict__ Bre, const float* __restrict__ Bim,
    const float* __restrict__ lstep, __bf16* __restrict__ Bmat,
    float2* __restrict__ lamb) {
  int idx = blockIdx.x * 256 + threadIdx.x;          // P*D
  int p = idx / Dsz, d = idx % Dsz;
  float lr = Lre[p], li = Lim[p];
  float dt = __expf(lstep[p]);
  float er = __expf(lr * dt);
  float lbr = er * __cosf(li * dt), lbi = er * __sinf(li * dt);
  float nr = lbr - 1.f, ni = lbi;
  float den = lr * lr + li * li;
  float cr = (nr * lr + ni * li) / den, ci = (ni * lr - nr * li) / den;
  float br = Bre[(size_t)p * Dsz + d], bi = Bim[(size_t)p * Dsz + d];
  Bmat[(size_t)(2 * p    ) * Dsz + d] = f2bf(cr * br - ci * bi);
  Bmat[(size_t)(2 * p + 1) * Dsz + d] = f2bf(cr * bi + ci * br);
  if (d == 0) lamb[p] = make_float2(lbr, lbi);
}

// ---------- Cmat bf16 (D rows, K=2P): [2*C_re | -2*C_im] interleaved -------
__global__ __launch_bounds__(256) void prep_cmat_kernel(
    const float* __restrict__ Cre, const float* __restrict__ Cim,
    __bf16* __restrict__ Cmat) {
  int idx = blockIdx.x * 256 + threadIdx.x;          // D*P
  int d = idx / Psz, p = idx % Psz;
  Cmat[(size_t)d * (2 * Psz) + 2 * p    ] = f2bf( 2.f * Cre[(size_t)d * Psz + p]);
  Cmat[(size_t)d * (2 * Psz) + 2 * p + 1] = f2bf(-2.f * Cim[(size_t)d * Psz + p]);
}

// ---------- LayerNorm + AdaLN modulate -> bf16 -----------------------------
__global__ __launch_bounds__(256) void ln_mod_a_kernel(
    const float* __restrict__ y, const float* __restrict__ mod,
    __bf16* __restrict__ u) {
  __shared__ float2 red[256];
  int row = blockIdx.x, b = row / Lsz, t = threadIdx.x;
  const float* yr = y + (size_t)row * Dsz;
  float v[4], s = 0.f, ss = 0.f;
  #pragma unroll
  for (int i = 0; i < 4; i++) { v[i] = yr[t + i * 256]; s += v[i]; ss += v[i] * v[i]; }
  red[t] = make_float2(s, ss); __syncthreads();
  for (int o = 128; o > 0; o >>= 1) {
    if (t < o) { red[t].x += red[t + o].x; red[t].y += red[t + o].y; }
    __syncthreads();
  }
  float mean = red[0].x / Dsz;
  float rstd = rsqrtf(red[0].y / Dsz - mean * mean + 1e-5f);
  const float* mb = mod + (size_t)b * 6 * Dsz;
  #pragma unroll
  for (int i = 0; i < 4; i++) {
    int d = t + i * 256;
    u[(size_t)row * Dsz + d] =
        f2bf((v[i] - mean) * rstd * (1.f + mb[Dsz + d]) + mb[d]);
  }
}

__global__ __launch_bounds__(256) void ln_mod_ff_kernel(
    const float* __restrict__ y1, const float* __restrict__ mod,
    const float* __restrict__ cond, __bf16* __restrict__ vout) {
  __shared__ float2 red[256];
  int row = blockIdx.x, b = row / Lsz, t = threadIdx.x;
  const float* yr = y1 + (size_t)row * Dsz;
  float v[4], s = 0.f, ss = 0.f;
  #pragma unroll
  for (int i = 0; i < 4; i++) { v[i] = yr[t + i * 256]; s += v[i]; ss += v[i] * v[i]; }
  red[t] = make_float2(s, ss); __syncthreads();
  for (int o = 128; o > 0; o >>= 1) {
    if (t < o) { red[t].x += red[t + o].x; red[t].y += red[t + o].y; }
    __syncthreads();
  }
  float mean = red[0].x / Dsz;
  float rstd = rsqrtf(red[0].y / Dsz - mean * mean + 1e-5f);
  const float* mb = mod + (size_t)b * 6 * Dsz;
  const float* cb = cond + (size_t)b * 2 * Dsz;
  #pragma unroll
  for (int i = 0; i < 4; i++) {
    int d = t + i * 256;
    float yn = (v[i] - mean) * rstd * (1.f + cb[d]) + cb[Dsz + d];
    vout[(size_t)row * Dsz + d] = f2bf(yn * (1.f + mb[4 * Dsz + d]) + mb[3 * Dsz + d]);
  }
}

// ---------- chunked complex associative scan -------------------------------
__global__ __launch_bounds__(256) void scan_p1_kernel(
    float* __restrict__ Bu, const float2* __restrict__ lamb,
    float2* __restrict__ carry) {
  int tid = blockIdx.x * 256 + threadIdx.x;          // B*NC*P
  int p = tid % Psz, rest = tid / Psz, c = rest % NC, b = rest / NC;
  float2 A = lamb[p];
  float2 s = make_float2(0.f, 0.f);
  float2* ptr = (float2*)(Bu + (size_t)(b * Lsz + c * CH) * (2 * Psz)) + p;
  for (int j = 0; j < CH; j++) {
    float2 x = ptr[(size_t)j * Psz];
    s = make_float2(A.x * s.x - A.y * s.y + x.x, A.x * s.y + A.y * s.x + x.y);
    ptr[(size_t)j * Psz] = s;
  }
  carry[tid] = s;
}

__global__ __launch_bounds__(256) void scan_p2_kernel(
    float2* __restrict__ carry, const float2* __restrict__ lamb) {
  int tid = blockIdx.x * 256 + threadIdx.x;          // B*P
  int p = tid % Psz, b = tid / Psz;
  float2 A = lamb[p], Ach = A;
  #pragma unroll
  for (int i = 0; i < 7; i++) Ach = cmul(Ach, Ach);  // A^128 (CH=2^7)
  float2 pre = make_float2(0.f, 0.f);
  for (int c = 0; c < NC; c++) {
    int idx = (b * NC + c) * Psz + p;
    float2 tmp = carry[idx];
    carry[idx] = pre;                                // state at chunk entry
    pre = make_float2(Ach.x * pre.x - Ach.y * pre.y + tmp.x,
                      Ach.x * pre.y + Ach.y * pre.x + tmp.y);
  }
}

__global__ __launch_bounds__(256) void scan_p3_kernel(
    const float* __restrict__ Bu, const float2* __restrict__ lamb,
    const float2* __restrict__ carry, unsigned* __restrict__ xs) {
  int tid = blockIdx.x * 256 + threadIdx.x;
  int p = tid % Psz, rest = tid / Psz, c = rest % NC, b = rest / NC;
  float2 A = lamb[p], s = carry[tid], pw = A;
  const float2* ptr = (const float2*)(Bu + (size_t)(b * Lsz + c * CH) * (2 * Psz)) + p;
  unsigned* xptr = xs + (size_t)(b * Lsz + c * CH) * Psz + p;
  for (int j = 0; j < CH; j++) {
    float2 loc = ptr[(size_t)j * Psz];
    float xr = loc.x + pw.x * s.x - pw.y * s.y;
    float xi = loc.y + pw.x * s.y + pw.y * s.x;
    pw = cmul(pw, A);
    union { __bf16 b2[2]; unsigned u; } pk;
    pk.b2[0] = f2bf(xr); pk.b2[1] = f2bf(xi);
    xptr[(size_t)j * Psz] = pk.u;
  }
}

// ---------- tiled WMMA bf16 GEMM, C = A @ Btᵀ, fp32 accumulate -------------
// A row-major (M,K) bf16; Bt is B transposed: (N,K) bf16, K contiguous.
// Global->LDS copies use GLOBAL_LOAD_ASYNC_TO_LDS_B128 (ASYNCcnt) with
// double-buffered LDS: one s_wait_asynccnt + one barrier per K-step.
constexpr int BM = 128, BN = 128, BK = 32, PK = 40;  // PK: padded halfs/row

__device__ inline void async_b128_to_lds(const __bf16* gptr, const __bf16* lptr) {
  unsigned lds = (unsigned)(uintptr_t)lptr;          // low 32b = LDS offset
  unsigned long long ga = (unsigned long long)(uintptr_t)gptr;
  asm volatile("global_load_async_to_lds_b128 %0, %1, off"
               :: "v"(lds), "v"(ga) : "memory");
}

template <int EPI>
__global__ __launch_bounds__(256) void gemm_bf16_kernel(
    const __bf16* __restrict__ A, int lda, const __bf16* __restrict__ Bt, int ldb,
    void* __restrict__ Cout, int ldc, int K, int m0,
    const float* __restrict__ e_y, const float* __restrict__ e_gate, int gstride,
    const float* __restrict__ e_vec, const __bf16* __restrict__ e_u) {
  __shared__ __bf16 As[2][BM * PK];
  __shared__ __bf16 Bs[2][BN * PK];
  const int tid = threadIdx.x, lane = tid & 31, wid = tid >> 5;
  const int wm = wid >> 2, wn = wid & 3;             // 2x4 wave grid
  const int l16 = lane & 15, lh = lane >> 4;
  const __bf16* Ab = A + (size_t)(blockIdx.y * BM) * lda;
  const __bf16* Bb = Bt + (size_t)(blockIdx.x * BN) * ldb;
  // this thread's two 16B chunks of the 128x32 tile
  const int r0 = (tid + 0)   >> 2, c0 = ((tid + 0)   & 3) * 8;
  const int r1 = (tid + 256) >> 2, c1 = ((tid + 256) & 3) * 8;

  v8f zero = {0.f, 0.f, 0.f, 0.f, 0.f, 0.f, 0.f, 0.f};
  v8f acc[4][2];
  #pragma unroll
  for (int s2 = 0; s2 < 4; s2++)
    #pragma unroll
    for (int t2 = 0; t2 < 2; t2++) acc[s2][t2] = zero;

  // prologue: async-load tile 0 into buffer 0
  async_b128_to_lds(&Ab[(size_t)r0 * lda + c0], &As[0][r0 * PK + c0]);
  async_b128_to_lds(&Ab[(size_t)r1 * lda + c1], &As[0][r1 * PK + c1]);
  async_b128_to_lds(&Bb[(size_t)r0 * ldb + c0], &Bs[0][r0 * PK + c0]);
  async_b128_to_lds(&Bb[(size_t)r1 * ldb + c1], &Bs[0][r1 * PK + c1]);

  for (int k0 = 0, it = 0; k0 < K; k0 += BK, ++it) {
    const int buf = it & 1;
    asm volatile("s_wait_asynccnt 0x0" ::: "memory"); // own async loads done
    __syncthreads();                                  // everyone's loads done,
                                                      // everyone done with buf^1
    if (k0 + BK < K) {                                // prefetch next tile
      const int kn = k0 + BK, nb = buf ^ 1;
      async_b128_to_lds(&Ab[(size_t)r0 * lda + kn + c0], &As[nb][r0 * PK + c0]);
      async_b128_to_lds(&Ab[(size_t)r1 * lda + kn + c1], &As[nb][r1 * PK + c1]);
      async_b128_to_lds(&Bb[(size_t)r0 * ldb + kn + c0], &Bs[nb][r0 * PK + c0]);
      async_b128_to_lds(&Bb[(size_t)r1 * ldb + kn + c1], &Bs[nb][r1 * PK + c1]);
    }
    union U { uint4 q[2]; v16bf v; };
    U bf[2];
    #pragma unroll
    for (int t2 = 0; t2 < 2; t2++) {                 // B 32x16: lane=col, ISA lane-half split
      int br = wn * 32 + t2 * 16 + l16;
      bf[t2].q[0] = *(uint4*)&Bs[buf][br * PK + lh * 8];
      bf[t2].q[1] = *(uint4*)&Bs[buf][br * PK + lh * 8 + 16];
    }
    #pragma unroll
    for (int s2 = 0; s2 < 4; s2++) {                 // A 16x32: ISA lane-half split
      U af;
      int ar = wm * 64 + s2 * 16 + l16;
      af.q[0] = *(uint4*)&As[buf][ar * PK + lh * 8];
      af.q[1] = *(uint4*)&As[buf][ar * PK + lh * 8 + 16];
      #pragma unroll
      for (int t2 = 0; t2 < 2; t2++)
        acc[s2][t2] = __builtin_amdgcn_wmma_f32_16x16x32_bf16(
            false, af.v, false, bf[t2].v, (short)0, acc[s2][t2], false, false);
    }
  }

  #pragma unroll
  for (int s2 = 0; s2 < 4; s2++)
    #pragma unroll
    for (int t2 = 0; t2 < 2; t2++) {
      int mb = blockIdx.y * BM + wm * 64 + s2 * 16 + lh * 8;
      int n  = blockIdx.x * BN + wn * 32 + t2 * 16 + l16;
      #pragma unroll
      for (int j = 0; j < 8; j++) {
        int m = mb + j;
        float val = acc[s2][t2][j];
        if constexpr (EPI == 0) {                    // raw f32 (Bu)
          ((float*)Cout)[(size_t)m * ldc + n] = val;
        } else if constexpr (EPI == 1) {             // y1 = y + gate_a*(a_y)
          int b = m / Lsz;
          float ay = val + e_vec[n] * bf2f(e_u[(size_t)m * Dsz + n]);
          ((float*)Cout)[(size_t)m * ldc + n] =
              e_y[(size_t)m * Dsz + n] + e_gate[(size_t)b * gstride + n] * ay;
        } else if constexpr (EPI == 2) {             // h = gelu(. + b1) -> bf16
          float x = val + e_vec[n];
          float g = 0.5f * x *
              (1.f + tanhf(0.7978845608028654f * (x + 0.044715f * x * x * x)));
          ((__bf16*)Cout)[(size_t)m * ldc + n] = f2bf(g);
        } else {                                     // out = y1 + gate_ff*(.+b2)
          int mabs = m0 + m, b = mabs / Lsz;
          ((float*)Cout)[(size_t)m * ldc + n] =
              e_y[(size_t)m * Dsz + n] +
              e_gate[(size_t)b * gstride + n] * (val + e_vec[n]);
        }
      }
    }
}

// ---------------------------------------------------------------------------
extern "C" void kernel_launch(void* const* d_in, const int* in_sizes, int n_in,
                              void* d_out, int out_size, void* d_ws, size_t ws_size,
                              hipStream_t stream) {
  (void)in_sizes; (void)n_in; (void)out_size; (void)ws_size;
  const float* y     = (const float*)d_in[0];
  const float* t     = (const float*)d_in[1];
  const float* Wm    = (const float*)d_in[2];
  const float* bm    = (const float*)d_in[3];
  const float* Lre   = (const float*)d_in[4];
  const float* Lim   = (const float*)d_in[5];
  const float* Bre   = (const float*)d_in[6];
  const float* Bim   = (const float*)d_in[7];
  const float* Cre   = (const float*)d_in[8];
  const float* Cim   = (const float*)d_in[9];
  const float* Ddiag = (const float*)d_in[10];
  const float* lstep = (const float*)d_in[11];
  const float* Wc    = (const float*)d_in[12];
  const float* bc    = (const float*)d_in[13];
  const float* W1    = (const float*)d_in[14];
  const float* b1    = (const float*)d_in[15];
  const float* W2    = (const float*)d_in[16];
  const float* b2    = (const float*)d_in[17];
  float* out = (float*)d_out;

  char* w = (char*)d_ws;
  const size_t MB = 1ull << 20;
  float*  mod   = (float*) (w + 0);            // 8*6144 f32      (192 KB)
  float*  cond  = (float*) (w + 256 * 1024);   // 8*2048 f32      (64 KB)
  float2* lamb  = (float2*)(w + 384 * 1024);   // 256 cplx        (2 KB)
  float2* carry = (float2*)(w + 512 * 1024);   // 32768 cplx      (256 KB)
  __bf16* Bmat  = (__bf16*)(w + 1 * MB);       // 512x1024 bf16   (1 MB)
  __bf16* Cmat  = (__bf16*)(w + 2 * MB);       // 1024x512 bf16   (1 MB)
  __bf16* W1T   = (__bf16*)(w + 4 * MB);       // 4096x1024 bf16  (8 MB)
  __bf16* W2T   = (__bf16*)(w + 12 * MB);      // 1024x4096 bf16  (8 MB)
  __bf16* u     = (__bf16*)(w + 20 * MB);      // 16384x1024 bf16 (32 MB)
  float*  Bu    = (float*) (w + 52 * MB);      // 16384x512 f32   (32 MB)
  __bf16* xs    = (__bf16*)(w + 84 * MB);      // 16384x512 bf16  (16 MB)
  float*  y1    = (float*) (w + 100 * MB);     // 16384x1024 f32  (64 MB)
  __bf16* v     = (__bf16*)(w + 52 * MB);      // alias Bu (dead after scan)
  __bf16* h     = (__bf16*)(w + 20 * MB);      // alias u (dead); 4096x4096 bf16

  // modulation / cond projections
  small_mm_kernel<<<dim3((8 * 6 * Dsz) / 256), 256, 0, stream>>>(t, Wm, bm, mod, 6 * Dsz, Dsz);
  small_mm_kernel<<<dim3((8 * 2 * Dsz) / 256), 256, 0, stream>>>(t, Wc, bc, cond, 2 * Dsz, Dsz);
  // ZOH discretization + weight conversion
  prep_bmat_kernel<<<dim3((Psz * Dsz) / 256), 256, 0, stream>>>(Lre, Lim, Bre, Bim, lstep, Bmat, lamb);
  prep_cmat_kernel<<<dim3((Dsz * Psz) / 256), 256, 0, stream>>>(Cre, Cim, Cmat);
  transpose_bf16_kernel<<<dim3(4 * Dsz / 32, Dsz / 32), dim3(32, 8), 0, stream>>>(W1, W1T, Dsz, 4 * Dsz);
  transpose_bf16_kernel<<<dim3(Dsz / 32, 4 * Dsz / 32), dim3(32, 8), 0, stream>>>(W2, W2T, 4 * Dsz, Dsz);
  // u = LN(y)*(1+scale_a)+shift_a   (bf16)
  ln_mod_a_kernel<<<Msz, 256, 0, stream>>>(y, mod, u);
  // Bu = u @ B_barᵀ  (complex-interleaved, f32)
  gemm_bf16_kernel<0><<<dim3((2 * Psz) / BN, Msz / BM), 256, 0, stream>>>(
      u, Dsz, Bmat, Dsz, Bu, 2 * Psz, Dsz, 0, nullptr, nullptr, 0, nullptr, nullptr);
  // 3-phase parallel scan over L
  scan_p1_kernel<<<dim3((Bsz * NC * Psz) / 256), 256, 0, stream>>>(Bu, lamb, carry);
  scan_p2_kernel<<<dim3((Bsz * Psz) / 256), 256, 0, stream>>>(carry, lamb);
  scan_p3_kernel<<<dim3((Bsz * NC * Psz) / 256), 256, 0, stream>>>(Bu, lamb, carry, (unsigned*)xs);
  // y1 = y + gate_a * (2*Re(xs@Cᵀ) + D_diag*u)
  gemm_bf16_kernel<1><<<dim3(Dsz / BN, Msz / BM), 256, 0, stream>>>(
      xs, 2 * Psz, Cmat, 2 * Psz, y1, Dsz, 2 * Psz, 0,
      y, mod + 2 * Dsz, 6 * Dsz, Ddiag, u);
  // v = modulate(AdaLN(y1)) (bf16)
  ln_mod_ff_kernel<<<Msz, 256, 0, stream>>>(y1, mod, cond, v);
  // FFN in 4 M-chunks to reuse workspace for h
  for (int c = 0; c < 4; c++) {
    const __bf16* vA = v + (size_t)c * 4096 * Dsz;
    gemm_bf16_kernel<2><<<dim3(4 * Dsz / BN, 4096 / BM), 256, 0, stream>>>(
        vA, Dsz, W1T, Dsz, h, 4 * Dsz, Dsz, 0,
        nullptr, nullptr, 0, b1, nullptr);
    gemm_bf16_kernel<3><<<dim3(Dsz / BN, 4096 / BM), 256, 0, stream>>>(
        h, 4 * Dsz, W2T, 4 * Dsz, out + (size_t)c * 4096 * Dsz, Dsz, 4 * Dsz,
        c * 4096, y1 + (size_t)c * 4096 * Dsz, mod + 5 * Dsz, 6 * Dsz, b2, nullptr);
  }
}
